// Attention_16389595202301
// MI455X (gfx1250) — matrix operand order repaired and checked by reading, not compiled, not run on hardware
//
#include <hip/hip_runtime.h>
#include <hip/hip_bf16.h>
#include <stdint.h>

typedef _Float16 f16;
typedef _Float16 v8h  __attribute__((ext_vector_type(8)));
typedef _Float16 v16h __attribute__((ext_vector_type(16)));
typedef float    v8f  __attribute__((ext_vector_type(8)));

#define H_  8
#define DH_ 64
#define N_  4096
#define D_  512
#define M_  256
#define CD_ 768
#define J_  4353          // N + 1 + M
#define JP_ 4384          // J padded to a multiple of 32

// Build a 16-half fragment from two 16-byte chunks (A- or B-fragment).
__device__ __forceinline__ v16h ld_frag(const f16* p0, const f16* p1) {
  v8h a = *(const v8h*)p0;
  v8h b = *(const v8h*)p1;
  v16h r;
#pragma unroll
  for (int i = 0; i < 8; ++i) { r[i] = a[i]; r[i + 8] = b[i]; }
  return r;
}

// ---------------------------------------------------------------- LayerNorm
template <typename OutT>
__global__ void ln_rows(const float* __restrict__ x, const float* __restrict__ sc,
                        const float* __restrict__ bs, OutT* __restrict__ out, int C) {
  __shared__ float red[256];
  const int row = blockIdx.x;
  const float* xr = x + (size_t)row * C;
  float s = 0.f;
  for (int c = threadIdx.x; c < C; c += 256) s += xr[c];
  red[threadIdx.x] = s; __syncthreads();
  for (int off = 128; off > 0; off >>= 1) {
    if (threadIdx.x < off) red[threadIdx.x] += red[threadIdx.x + off];
    __syncthreads();
  }
  const float mu = red[0] / C;
  __syncthreads();
  float v = 0.f;
  for (int c = threadIdx.x; c < C; c += 256) { float d = xr[c] - mu; v += d * d; }
  red[threadIdx.x] = v; __syncthreads();
  for (int off = 128; off > 0; off >>= 1) {
    if (threadIdx.x < off) red[threadIdx.x] += red[threadIdx.x + off];
    __syncthreads();
  }
  const float rstd = rsqrtf(red[0] / C + 1e-6f);
  for (int c = threadIdx.x; c < C; c += 256)
    out[(size_t)row * C + c] = (OutT)((xr[c] - mu) * rstd * sc[c] + bs[c]);
}

// ------------------------------------------- weight transpose f32 -> f16
__global__ void transpose_to_f16(const float* __restrict__ in, f16* __restrict__ out,
                                 int R, int C) {
  size_t idx = (size_t)blockIdx.x * blockDim.x + threadIdx.x;
  if (idx >= (size_t)R * C) return;
  int r = (int)(idx / C), c = (int)(idx % C);
  out[(size_t)c * R + r] = (f16)in[idx];
}

// ------------------- null-KV row, zero padding, additive mask vector
__global__ void prep_kv(const float* __restrict__ null_kv, const uint8_t* __restrict__ mask,
                        f16* __restrict__ K16, f16* __restrict__ Vt, float* __restrict__ maskJ) {
  int tid = blockIdx.x * blockDim.x + threadIdx.x;
  if (tid < DH_) {                                   // null kv -> j = N_
    K16[(size_t)N_ * DH_ + tid] = (f16)null_kv[tid];
    Vt[(size_t)tid * JP_ + N_]  = (f16)null_kv[DH_ + tid];
  }
  if (tid < JP_) {                                   // additive mask over j
    float v;
    if (tid == 0)       v = 0.f;                     // prepended null slot: True
    else if (tid < J_)  v = mask[tid - 1] ? 0.f : -1e30f;
    else                v = -1e30f;                  // padding
    maskJ[tid] = v;
  }
  if (tid >= J_ && tid < JP_) {                      // zero K/V padding rows
    for (int d = 0; d < DH_; ++d) {
      K16[(size_t)tid * DH_ + d] = (f16)0.f;
      Vt[(size_t)d * JP_ + tid]  = (f16)0.f;
    }
  }
}

// ---------------------------------------------------------- WMMA GEMM
// C[i0..i0+15][n0..n0+63] = A(M x K, f16 row-major) @ Bt^T  (Bt is N x K f16)
// mode 0: out16 = alpha * c; mode 1: split KV (K row-major / Vt transposed); mode 2: f32 out
__global__ void gemm_wmma(const f16* __restrict__ A, const f16* __restrict__ Bt, int K,
                          float alpha, const float* __restrict__ bias,
                          f16* __restrict__ out16, int ldo,
                          f16* __restrict__ outK, f16* __restrict__ outVt, int ldVt, int rowOff,
                          float* __restrict__ outF, int mode) {
  const int lane = threadIdx.x;
  const int col = lane & 15, hl = lane >> 4;
  const int i0 = blockIdx.x * 16;
  const int n0 = blockIdx.y * 64;

  v8f c[4] = {};
  for (int kc = 0; kc < K; kc += 32) {
    const f16* ap = A + (size_t)(i0 + col) * K + kc + hl * 8;
    v16h a = ld_frag(ap, ap + 16);
#pragma unroll
    for (int t = 0; t < 4; ++t) {
      const f16* bp = Bt + (size_t)(n0 + t * 16 + col) * K + kc + hl * 8;
      v16h b = ld_frag(bp, bp + 16);
      c[t] = __builtin_amdgcn_wmma_f32_16x16x32_f16(false, a, false, b, (short)0,
                                                    c[t], false, false);
    }
  }
#pragma unroll
  for (int t = 0; t < 4; ++t) {
    const int n = n0 + t * 16 + col;
#pragma unroll
    for (int r = 0; r < 8; ++r) {
      const int row = i0 + r + 8 * hl;
      float v = c[t][r];
      if (mode == 0) {
        out16[(size_t)row * ldo + n] = (f16)(v * alpha);
      } else if (mode == 1) {
        if (bias) v += bias[n];
        if (n < DH_) outK[(size_t)(row + rowOff) * DH_ + n] = (f16)v;
        else         outVt[(size_t)(n - DH_) * ldVt + (row + rowOff)] = (f16)v;
      } else {
        outF[(size_t)row * ldo + n] = v;
      }
    }
  }
}

// ------------------------------------------- fused flash attention core
// grid = (N/64, H), block = 128 (4 waves; each wave owns a 16-row Q tile).
// Per 32-key step, the workgroup stages the shared K tile (4KB) and V tile (4KB)
// into LDS with gfx1250 async global->LDS copies (ASYNCcnt), then all waves
// consume them via ds_load fragments feeding v_wmma.
__global__ void flash_attn(const f16* __restrict__ q16, const f16* __restrict__ K16,
                           const f16* __restrict__ Vt, const float* __restrict__ bias,
                           const float* __restrict__ maskJ, f16* __restrict__ O16) {
  __shared__ __align__(16) f16 Kt[32 * 64];     // [j-local][d]
  __shared__ __align__(16) f16 Vtile[64 * 32];  // [d][j-local]
  __shared__ __align__(16) f16 P[4][16 * 32];   // per-wave P staging

  const int tid  = threadIdx.x;
  const int wave = tid >> 5, lane = tid & 31;
  const int col = lane & 15, hl = (lane >> 4) & 1;
  const int i0 = blockIdx.x * 64 + wave * 16;
  const int h  = blockIdx.y;

  const uint32_t ldsK = (uint32_t)(uintptr_t)(&Kt[0]);
  const uint32_t ldsV = (uint32_t)(uintptr_t)(&Vtile[0]);

  // Q A-fragments (16 rows x d-chunks {0..31, 32..63}), held for the whole loop.
  const f16* qp = q16 + (size_t)(i0 + col) * D_ + h * DH_ + hl * 8;
  const v16h a0 = ld_frag(qp, qp + 16);
  const v16h a1 = ld_frag(qp + 32, qp + 48);

  float mrow[8], lrow[8];
  v8f o[4] = {};
#pragma unroll
  for (int r = 0; r < 8; ++r) { mrow[r] = -1e30f; lrow[r] = 0.f; }

  for (int jc = 0; jc < JP_; jc += 32) {
    // ---- async stage K tile (rows jc..jc+31, contiguous 4KB) and V tile ----
    {
      const f16* gK = K16 + (size_t)jc * DH_;   // contiguous block
      const f16* gV = Vt + jc;                  // 64 rows, 64B each, stride JP_*2
#pragma unroll
      for (int s = 0; s < 2; ++s) {
        const int c = tid + s * 128;            // 16-byte chunk id, 0..255
        asm volatile("global_load_async_to_lds_b128 %0, %1, %2"
                     :: "v"(ldsK + c * 16), "v"(c * 16), "s"(gK) : "memory");
        const int d = c >> 2, jl = (c & 3) * 8;
        asm volatile("global_load_async_to_lds_b128 %0, %1, %2"
                     :: "v"(ldsV + c * 16), "v"((d * JP_ + jl) * 2), "s"(gV) : "memory");
      }
    }

    // ---- bias + mask -> WMMA C init (overlaps the async copies) ----
    const int j0 = jc + col, j1 = jc + 16 + col;
    const float mj0 = maskJ[j0], mj1 = maskJ[j1];
    v8f c0, c1;
#pragma unroll
    for (int r = 0; r < 8; ++r) {
      const int row = i0 + r + 8 * hl;
      const size_t base = ((size_t)h * N_ + row) * J_;
      c0[r] = ((j0 < J_) ? bias[base + j0] : 0.f) + mj0;
      c1[r] = ((j1 < J_) ? bias[base + j1] : 0.f) + mj1;
    }

    asm volatile("s_wait_asynccnt 0x0" ::: "memory");
    __syncthreads();

    // ---- K B-fragments from LDS: lane = key j-local, d in table order per lane ----
    const f16* k0 = &Kt[(size_t)col * DH_ + hl * 8];
    const f16* k1 = &Kt[(size_t)(col + 16) * DH_ + hl * 8];
    v16h b00 = ld_frag(k0, k0 + 16),  b01 = ld_frag(k0 + 32, k0 + 48);
    v16h b10 = ld_frag(k1, k1 + 16),  b11 = ld_frag(k1 + 32, k1 + 48);

    c0 = __builtin_amdgcn_wmma_f32_16x16x32_f16(false, a0, false, b00, (short)0, c0, false, false);
    c0 = __builtin_amdgcn_wmma_f32_16x16x32_f16(false, a1, false, b01, (short)0, c0, false, false);
    c1 = __builtin_amdgcn_wmma_f32_16x16x32_f16(false, a0, false, b10, (short)0, c1, false, false);
    c1 = __builtin_amdgcn_wmma_f32_16x16x32_f16(false, a1, false, b11, (short)0, c1, false, false);

    // ---- online softmax: reduce across the 16 lanes that share a row ----
    float f[8];
#pragma unroll
    for (int r = 0; r < 8; ++r) {
      float v = fmaxf(c0[r], c1[r]);
      v = fmaxf(v, __shfl_xor(v, 1, 32));
      v = fmaxf(v, __shfl_xor(v, 2, 32));
      v = fmaxf(v, __shfl_xor(v, 4, 32));
      v = fmaxf(v, __shfl_xor(v, 8, 32));
      const float mn = fmaxf(mrow[r], v);
      f[r] = __expf(mrow[r] - mn);
      mrow[r] = mn;
      const float p0 = __expf(c0[r] - mn);
      const float p1 = __expf(c1[r] - mn);
      c0[r] = p0; c1[r] = p1;
      float rs = p0 + p1;
      rs += __shfl_xor(rs, 1, 32);
      rs += __shfl_xor(rs, 2, 32);
      rs += __shfl_xor(rs, 4, 32);
      rs += __shfl_xor(rs, 8, 32);
      lrow[r] = lrow[r] * f[r] + rs;
    }
#pragma unroll
    for (int t = 0; t < 4; ++t)
#pragma unroll
      for (int r = 0; r < 8; ++r) o[t][r] *= f[r];

    // ---- reshape P tile (C-layout -> A-layout) via per-wave LDS region ----
    f16* Pw = P[wave];
#pragma unroll
    for (int r = 0; r < 8; ++r) {
      Pw[(r + 8 * hl) * 32 + col]      = (f16)c0[r];
      Pw[(r + 8 * hl) * 32 + 16 + col] = (f16)c1[r];
    }
    const f16* pp = Pw + (size_t)col * 32 + hl * 8;
    const v16h pa = ld_frag(pp, pp + 16);

    // ---- P @ V from the shared LDS V tile ----
#pragma unroll
    for (int t = 0; t < 4; ++t) {
      const f16* vp = &Vtile[(size_t)(t * 16 + col) * 32 + hl * 8];
      v16h vb = ld_frag(vp, vp + 16);
      o[t] = __builtin_amdgcn_wmma_f32_16x16x32_f16(false, pa, false, vb, (short)0,
                                                    o[t], false, false);
    }

    __syncthreads();   // all waves done reading Kt/Vtile before next async stage
  }

  // normalize and write O in (n, h*64+d) layout for the output GEMM
#pragma unroll
  for (int r = 0; r < 8; ++r) {
    const float inv = 1.f / lrow[r];
    const int row = i0 + r + 8 * hl;
#pragma unroll
    for (int t = 0; t < 4; ++t)
      O16[(size_t)row * D_ + h * DH_ + t * 16 + col] = (f16)(o[t][r] * inv);
  }
}

// ---------------------------------------------------------------- launch
extern "C" void kernel_launch(void* const* d_in, const int* in_sizes, int n_in,
                              void* d_out, int out_size, void* d_ws, size_t ws_size,
                              hipStream_t stream) {
  (void)in_sizes; (void)n_in; (void)out_size; (void)ws_size;
  const float*   x        = (const float*)d_in[0];
  const float*   context  = (const float*)d_in[1];
  const float*   abias    = (const float*)d_in[2];
  const uint8_t* mask     = (const uint8_t*)d_in[3];
  const float*   ln1_s    = (const float*)d_in[4];
  const float*   ln1_b    = (const float*)d_in[5];
  const float*   wq       = (const float*)d_in[6];
  const float*   wkv      = (const float*)d_in[7];
  const float*   null_kv  = (const float*)d_in[8];
  const float*   lnc_s    = (const float*)d_in[9];
  const float*   lnc_b    = (const float*)d_in[10];
  const float*   wckv     = (const float*)d_in[11];
  const float*   bckv     = (const float*)d_in[12];
  const float*   wo       = (const float*)d_in[13];
  const float*   lno_s    = (const float*)d_in[14];
  const float*   lno_b    = (const float*)d_in[15];
  float* out = (float*)d_out;

  char* w = (char*)d_ws;
  size_t off = 0;
  auto alloc = [&](size_t bytes) { char* p = w + off; off = (off + bytes + 255) & ~(size_t)255; return p; };
  f16*   xh16    = (f16*)alloc((size_t)N_ * D_ * 2);
  f16*   ch16    = (f16*)alloc((size_t)M_ * CD_ * 2);
  f16*   wqT     = (f16*)alloc((size_t)D_ * D_ * 2);
  f16*   wkvT    = (f16*)alloc((size_t)128 * D_ * 2);
  f16*   wckvT   = (f16*)alloc((size_t)128 * CD_ * 2);
  f16*   woT     = (f16*)alloc((size_t)D_ * D_ * 2);
  f16*   q16     = (f16*)alloc((size_t)N_ * D_ * 2);
  f16*   K16     = (f16*)alloc((size_t)JP_ * DH_ * 2);
  f16*   Vt16    = (f16*)alloc((size_t)DH_ * JP_ * 2);
  float* maskJ   = (float*)alloc((size_t)JP_ * 4);
  f16*   O16     = (f16*)alloc((size_t)N_ * D_ * 2);
  float* outF32  = (float*)alloc((size_t)N_ * D_ * 4);

  // 1) LayerNorms -> f16
  ln_rows<f16><<<N_, 256, 0, stream>>>(x, ln1_s, ln1_b, xh16, D_);
  ln_rows<f16><<<M_, 256, 0, stream>>>(context, lnc_s, lnc_b, ch16, CD_);

  // 2) Weight transposes (f32 -> f16, N x K layout for B-fragments)
  transpose_to_f16<<<(D_ * D_ + 255) / 256, 256, 0, stream>>>(wq, wqT, D_, D_);
  transpose_to_f16<<<(D_ * 128 + 255) / 256, 256, 0, stream>>>(wkv, wkvT, D_, 128);
  transpose_to_f16<<<(CD_ * 128 + 255) / 256, 256, 0, stream>>>(wckv, wckvT, CD_, 128);
  transpose_to_f16<<<(D_ * D_ + 255) / 256, 256, 0, stream>>>(wo, woT, D_, D_);

  // 3) null-KV row, mask vector, K/V padding
  prep_kv<<<(JP_ + 255) / 256, 256, 0, stream>>>(null_kv, mask, K16, Vt16, maskJ);

  // 4) q = LN(x) @ wq * DH^-0.5
  gemm_wmma<<<dim3(N_ / 16, D_ / 64), 32, 0, stream>>>(
      xh16, wqT, D_, 0.125f, nullptr, q16, D_, nullptr, nullptr, 0, 0, nullptr, 0);

  // 5) self k,v = LN(x) @ wkv  -> K rows [0,4096), Vt cols [0,4096)
  gemm_wmma<<<dim3(N_ / 16, 128 / 64), 32, 0, stream>>>(
      xh16, wkvT, D_, 1.f, nullptr, nullptr, 0, K16, Vt16, JP_, 0, nullptr, 1);

  // 6) context k,v = LN(ctx) @ wckv + bckv -> rows/cols [4097, 4353)
  gemm_wmma<<<dim3(M_ / 16, 128 / 64), 32, 0, stream>>>(
      ch16, wckvT, CD_, 1.f, bckv, nullptr, 0, K16, Vt16, JP_, N_ + 1, nullptr, 1);

  // 7) fused attention (streams attn_bias once; async K/V staging into LDS)
  flash_attn<<<dim3(N_ / 64, H_), 128, 0, stream>>>(q16, K16, Vt16, abias, maskJ, O16);

  // 8) out = O @ wo  (f32)
  gemm_wmma<<<dim3(N_ / 16, D_ / 64), 32, 0, stream>>>(
      O16, woT, D_, 1.f, nullptr, nullptr, 0, nullptr, nullptr, 0, 0, outF32, 2);

  // 9) final LayerNorm -> d_out (f32)
  ln_rows<float><<<N_, 256, 0, stream>>>(outF32, lno_s, lno_b, out, D_);
}
